// TrellisLinear_20306605375936
// MI455X (gfx1250) — matrix operand order repaired and checked by reading, not compiled, not run on hardware
//
#include <hip/hip_runtime.h>

// TrellisLinear for MI455X (gfx1250, wave32, WMMA).
// out(4096x4096) = X(4096x4096) @ W(4096x4096)^T, W built from trellis decode
// + row FWHT*SU + col FWHT*SV + block scales.  GEMM runs on
// v_wmma_f32_16x16x32_bf16 with 64x64 wave tiles and double-buffered LDS.

typedef __attribute__((ext_vector_type(16))) __bf16 v16bf;
typedef __attribute__((ext_vector_type(8)))  __bf16 v8bf;
typedef __attribute__((ext_vector_type(8)))  float  v8f;

union FragBF16 { v16bf v; v8bf h[2]; };

// ---------------------------------------------------------------------------
// Kernel 1: trellis decode + 4096-pt row FWHT + SU  ->  W1 (f32, row major)
// one workgroup (256 thr) per output row; 16 KB LDS
// ---------------------------------------------------------------------------
__global__ __launch_bounds__(256)
void k_decode_rowfwht(const int* __restrict__ trellis,
                      const float* __restrict__ tlut,
                      const float* __restrict__ SU,
                      float* __restrict__ W1)
{
    __shared__ float row[4096];
    const int o   = blockIdx.x;
    const int tid = threadIdx.x;

    // ---- decode: 16 trellis rows (32 words) -> 256 values each ----
    const int j = tid >> 4;          // trellis row within this W row
    const int q = tid & 15;          // this thread handles 8 states
    const int* tr = trellis + (o * 16 + j) * 32;
#pragma unroll
    for (int k = 0; k < 8; ++k) {
        int s  = q * 8 + k;          // state index 0..127
        int i0 = s >> 2;
        int sh = (s & 3) * 4;
        int i1 = (i0 + 1) & 31;
        unsigned w0 = (unsigned)tr[i0];
        unsigned w1 = (unsigned)tr[i1];
        unsigned st = ((w0 << sh) | (w1 >> (16 - sh))) & 0xFFFFu;
        int idx = (int)(st & 1023u);
        row[j * 256 + s * 2 + 0] = tlut[idx * 2 + 0];
        row[j * 256 + s * 2 + 1] = tlut[idx * 2 + 1];
    }
    __syncthreads();

    // ---- 4096-pt FWHT: 12 stages, 2048 butterflies/stage, 8 per thread ----
    for (int h = 1; h < 4096; h <<= 1) {
#pragma unroll
        for (int k = 0; k < 8; ++k) {
            int b = tid + k * 256;
            int i = ((b & ~(h - 1)) << 1) | (b & (h - 1));
            float a = row[i];
            float c = row[i + h];
            row[i]     = a + c;
            row[i + h] = a - c;
        }
        __syncthreads();
    }

    // normalize 1/sqrt(4096)=1/64, apply SU, write out
#pragma unroll
    for (int k = 0; k < 16; ++k) {
        int d = tid + k * 256;
        W1[(size_t)o * 4096 + d] = row[d] * 0.015625f * SU[d];
    }
}

// ---------------------------------------------------------------------------
// Kernel 2a: column FWHT stage 1 (H64 over o_lo), in place on W1.
// H4096 = H64 (x) H64 tensor factorization; 64x64 f32 tiles in LDS.
// ---------------------------------------------------------------------------
__global__ __launch_bounds__(256)
void k_colfwht_lo(float* __restrict__ W1)
{
    __shared__ float tile[64 * 64];
    const int ct  = blockIdx.x;              // 64-column tile
    const int ohi = blockIdx.y;              // contiguous row group
    const int tid = threadIdx.x;
    const int r   = tid >> 2;
    const int c0  = (tid & 3) * 16;

    const size_t gbase = (size_t)(ohi * 64 + r) * 4096 + ct * 64 + c0;
    {
        const float4* src = (const float4*)(W1 + gbase);
        float4* dst = (float4*)(tile + r * 64 + c0);
        dst[0] = src[0]; dst[1] = src[1]; dst[2] = src[2]; dst[3] = src[3];
    }
    __syncthreads();

    for (int h = 1; h < 64; h <<= 1) {
#pragma unroll
        for (int k = 0; k < 8; ++k) {
            int b   = tid + k * 256;         // 2048 butterflies
            int col = b & 63;
            int p   = b >> 6;                // 0..31
            int i   = ((p & ~(h - 1)) << 1) | (p & (h - 1));
            float a = tile[i * 64 + col];
            float c = tile[(i + h) * 64 + col];
            tile[i * 64 + col]       = a + c;
            tile[(i + h) * 64 + col] = a - c;
        }
        __syncthreads();
    }

    float4* dst = (float4*)(W1 + gbase);
    const float4* src = (const float4*)(tile + r * 64 + c0);
#pragma unroll
    for (int i = 0; i < 4; ++i) {
        float4 v = src[i];
        v.x *= 0.125f; v.y *= 0.125f; v.z *= 0.125f; v.w *= 0.125f;  // 1/sqrt(64)
        dst[i] = v;
    }
}

// ---------------------------------------------------------------------------
// Kernel 2b: column FWHT stage 2 (H64 over o_hi, stride-64 rows) + SV +
// per-256-block scales -> Wbf (bf16, row major OUTxIN)
// ---------------------------------------------------------------------------
__global__ __launch_bounds__(256)
void k_colfwht_hi(const float* __restrict__ W1,
                  const float* __restrict__ SV,
                  const float* __restrict__ scales,
                  __bf16* __restrict__ Wbf)
{
    __shared__ float tile[64 * 64];
    const int ct  = blockIdx.x;
    const int olo = blockIdx.y;
    const int tid = threadIdx.x;
    const int r   = tid >> 2;                // o_hi index
    const int c0  = (tid & 3) * 16;
    const int orow = r * 64 + olo;

    {
        const float4* src = (const float4*)(W1 + (size_t)orow * 4096 + ct * 64 + c0);
        float4* dst = (float4*)(tile + r * 64 + c0);
        dst[0] = src[0]; dst[1] = src[1]; dst[2] = src[2]; dst[3] = src[3];
    }
    __syncthreads();

    for (int h = 1; h < 64; h <<= 1) {
#pragma unroll
        for (int k = 0; k < 8; ++k) {
            int b   = tid + k * 256;
            int col = b & 63;
            int p   = b >> 6;
            int i   = ((p & ~(h - 1)) << 1) | (p & (h - 1));
            float a = tile[i * 64 + col];
            float c = tile[(i + h) * 64 + col];
            tile[i * 64 + col]       = a + c;
            tile[(i + h) * 64 + col] = a - c;
        }
        __syncthreads();
    }

    // scale: 1/sqrt(64) * SV[o] * scales[o*16 + d/256]  (64-col tile stays in
    // one 256-block since ct*64..ct*64+63 >> 8 is constant)
    const float s = 0.125f * SV[orow] * scales[orow * 16 + (ct >> 2)];
    __bf16* dst = Wbf + (size_t)orow * 4096 + ct * 64 + c0;
    const float* t = tile + r * 64 + c0;
#pragma unroll
    for (int i = 0; i < 16; ++i) dst[i] = (__bf16)(t[i] * s);
}

// ---------------------------------------------------------------------------
// Kernel 3: GEMM  out[m,n] = sum_k X[m,k] * W[n,k]   (bf16 WMMA, f32 acc)
// block tile 128x256, 8 waves (2 M x 4 N) of 64x64 (4x4 wmma accumulators),
// BK=32, double-buffered LDS with 80B row pitch (conflict-free b128 reads).
// ---------------------------------------------------------------------------
#define LPITCH 40   // bf16 elements per LDS row (80 bytes)

__global__ __launch_bounds__(256)
void k_gemm(const float* __restrict__ X,
            const __bf16* __restrict__ Wbf,
            float* __restrict__ Out)
{
    __shared__ __align__(16) __bf16 lX[2][128 * LPITCH];   // 2 x 10 KB
    __shared__ __align__(16) __bf16 lW[2][256 * LPITCH];   // 2 x 20 KB

    const int tid  = threadIdx.x;
    const int lane = tid & 31;
    const int wave = tid >> 5;          // 0..7
    const int wm   = wave & 1;          // M slot (x64)
    const int wn   = wave >> 1;         // N slot (x64)
    const int m0   = blockIdx.y * 128;
    const int n0   = blockIdx.x * 256;

    const int xrow = tid >> 1;          // X staging row 0..127
    const int xseg = tid & 1;           // 16-float segment in K
    const int wrow = tid;               // W staging row 0..255 (32 bf16 each)

    const float* gX = X   + (size_t)(m0 + xrow) * 4096 + xseg * 16;
    const __bf16* gW = Wbf + (size_t)(n0 + wrow) * 4096;

    v8f acc[4][4];
#pragma unroll
    for (int mi = 0; mi < 4; ++mi)
#pragma unroll
        for (int ni = 0; ni < 4; ++ni)
            acc[mi][ni] = (v8f)(0.0f);

    const int half = lane >> 4;
    const int lrow = lane & 15;

    float4 xr[4];
    uint4  wr[4];

    // ---- prologue: fetch k-tile 0 and stage into buffer 0 ----
    {
        const float4* g = (const float4*)gX;
        xr[0] = g[0]; xr[1] = g[1]; xr[2] = g[2]; xr[3] = g[3];
        const uint4* w = (const uint4*)gW;
        wr[0] = w[0]; wr[1] = w[1]; wr[2] = w[2]; wr[3] = w[3];
    }
    {
        __bf16* d = &lX[0][xrow * LPITCH + xseg * 16];
        const float* f = (const float*)xr;
#pragma unroll
        for (int i = 0; i < 16; ++i) d[i] = (__bf16)f[i];
        uint4* dw = (uint4*)&lW[0][wrow * LPITCH];
        dw[0] = wr[0]; dw[1] = wr[1]; dw[2] = wr[2]; dw[3] = wr[3];
    }
    __syncthreads();

    for (int kt = 0; kt < 128; ++kt) {
        const int buf = kt & 1;
        const bool more = (kt + 1) < 128;

        // ---- issue global loads for the next k-tile (overlap with compute)
        if (more) {
            const float4* g = (const float4*)(gX + (kt + 1) * 32);
            xr[0] = g[0]; xr[1] = g[1]; xr[2] = g[2]; xr[3] = g[3];
            const uint4* w = (const uint4*)(gW + (kt + 1) * 32);
            wr[0] = w[0]; wr[1] = w[1]; wr[2] = w[2]; wr[3] = w[3];
        }
        if (kt + 2 < 128) {
            __builtin_prefetch(gX + (kt + 2) * 32, 0, 0);
            __builtin_prefetch(gW + (kt + 2) * 32, 0, 0);
        }

        // ---- compute from lds[buf]: 16 wmma per wave ----
        // B (32x16): lane<16 -> col n, K0..15 ; lane>=16 -> col n, K16..31
        FragBF16 b[4];
#pragma unroll
        for (int ni = 0; ni < 4; ++ni) {
            const v8bf* p = (const v8bf*)&lW[buf][(wn * 64 + ni * 16 + lrow) * LPITCH];
            b[ni].h[0] = p[2 * half];
            b[ni].h[1] = p[2 * half + 1];
        }
        // A (16x32): lane<16 -> row m, K0..15 ; lane>=16 -> row m, K16..31
#pragma unroll
        for (int mi = 0; mi < 4; ++mi) {
            FragBF16 a;
            const v8bf* p = (const v8bf*)&lX[buf][(wm * 64 + mi * 16 + lrow) * LPITCH];
            a.h[0] = p[2 * half];
            a.h[1] = p[2 * half + 1];
#pragma unroll
            for (int ni = 0; ni < 4; ++ni)
                acc[mi][ni] = __builtin_amdgcn_wmma_f32_16x16x32_bf16(
                    false, a.v, false, b[ni].v,
                    (short)0, acc[mi][ni], false, false);
        }

        // ---- stage next tile into the other buffer, single barrier ----
        if (more) {
            __bf16* d = &lX[1 - buf][xrow * LPITCH + xseg * 16];
            const float* f = (const float*)xr;
#pragma unroll
            for (int i = 0; i < 16; ++i) d[i] = (__bf16)f[i];
            uint4* dw = (uint4*)&lW[1 - buf][wrow * LPITCH];
            dw[0] = wr[0]; dw[1] = wr[1]; dw[2] = wr[2]; dw[3] = wr[3];
        }
        __syncthreads();
    }

    // ---- epilogue: C/D layout = VGPR v -> row v + 8*(lane>=16), col = lane&15
#pragma unroll
    for (int mi = 0; mi < 4; ++mi) {
#pragma unroll
        for (int ni = 0; ni < 4; ++ni) {
            int n     = n0 + wn * 64 + ni * 16 + (lane & 15);
            int mbase = m0 + wm * 64 + mi * 16 + ((lane >> 4) << 3);
#pragma unroll
            for (int v = 0; v < 8; ++v)
                Out[(size_t)(mbase + v) * 4096 + n] = acc[mi][ni][v];
        }
    }
}

// ---------------------------------------------------------------------------
extern "C" void kernel_launch(void* const* d_in, const int* in_sizes, int n_in,
                              void* d_out, int out_size, void* d_ws, size_t ws_size,
                              hipStream_t stream)
{
    (void)in_sizes; (void)n_in; (void)out_size; (void)ws_size;

    const float* x       = (const float*)d_in[0];   // (4,1024,4096) f32
    const float* SU      = (const float*)d_in[1];   // (4096,)
    const float* SV      = (const float*)d_in[2];   // (4096,)
    const float* scales  = (const float*)d_in[3];   // (65536,)
    const float* tlut    = (const float*)d_in[4];   // (1024,2)
    const int*   trellis = (const int*)d_in[5];     // (65536,32) int32
    float* out = (float*)d_out;                     // (4,1024,4096) f32

    // workspace: W1 f32 (64 MB) | Wbf bf16 (32 MB)
    float*  W1  = (float*)d_ws;
    __bf16* Wbf = (__bf16*)((char*)d_ws + (size_t)4096 * 4096 * sizeof(float));

    k_decode_rowfwht<<<4096, 256, 0, stream>>>(trellis, tlut, SU, W1);
    k_colfwht_lo   <<<dim3(64, 64), 256, 0, stream>>>(W1);
    k_colfwht_hi   <<<dim3(64, 64), 256, 0, stream>>>(W1, SV, scales, Wbf);
    k_gemm         <<<dim3(16, 32), 256, 0, stream>>>(x, Wbf, out);
}
